// Attention_77584289235090
// MI455X (gfx1250) — compile-verified, matrix operand order checked by reading, not run on hardware
//
#include <hip/hip_runtime.h>
#include <hip/hip_bf16.h>

typedef __attribute__((ext_vector_type(16))) __bf16 v16bf;
typedef __attribute__((ext_vector_type(8)))  float  v8f;

#define T_SEQ   4096
#define D_DIM   1024
#define H_DIM   256
#define B_BATCH 32
#define NEG_BIG -99999999.0f

static __device__ __forceinline__ unsigned short bfbits(float f) {
    __bf16 b = (__bf16)f;
    return __builtin_bit_cast(unsigned short, b);
}

// ---------------------------------------------------------------------------
// Kernel 1: convert W1 (fp32, DxH row-major) into bf16 WMMA B-fragment layout.
// Fragment (kc, nt) holds W1[kc*32 .. kc*32+31, nt*16 .. nt*16+15] as the
// 32x16 bf16 B operand: lane n<16 -> col N=n, K in {0..7,16..23};
// lane>=16 -> col N=n-16, K in {8..15,24..31}. Each lane's 16 bf16 are
// stored contiguously (32B). Frags sequential: idx = kc*16+nt, 1KB each.
// ---------------------------------------------------------------------------
__global__ void __launch_bounds__(256) conv_w1_kernel(
    const float* __restrict__ W1, unsigned int* __restrict__ w1sw) {
    int gid  = blockIdx.x * 256 + threadIdx.x;   // 0..16383
    int lane = gid & 31;
    int frag = gid >> 5;                         // 0..511
    int kc   = frag >> 4;                        // 0..31
    int nt   = frag & 15;                        // 0..15
    int n    = lane & 15;
    int h    = nt * 16 + n;
    int kbase = kc * 32 + ((lane & 16) ? 8 : 0);

    unsigned int pk[8];
#pragma unroll
    for (int j = 0; j < 8; ++j) {
        int i0 = 2 * j, i1 = 2 * j + 1;
        int k0 = kbase + (i0 < 8 ? i0 : 16 + (i0 - 8));
        int k1 = kbase + (i1 < 8 ? i1 : 16 + (i1 - 8));
        unsigned int lo = bfbits(W1[(size_t)k0 * H_DIM + h]);
        unsigned int hi = bfbits(W1[(size_t)k1 * H_DIM + h]);
        pk[j] = lo | (hi << 16);
    }
    unsigned int* dst = w1sw + (size_t)frag * 256 + lane * 8;
    ((uint4*)dst)[0] = make_uint4(pk[0], pk[1], pk[2], pk[3]);
    ((uint4*)dst)[1] = make_uint4(pk[4], pk[5], pk[6], pk[7]);
}

// ---------------------------------------------------------------------------
// Kernel 2: fused GEMM + tanh + dot(W2): e[b,t] for all 131072 rows.
// Block = 256 threads = 8 waves; each wave owns a 16-row tile of x.
// Per K-chunk (32 wide): stage 16 B-fragments (16KB) into LDS with
// GLOBAL_LOAD_ASYNC_TO_LDS_B128 (ASYNCcnt path, no VGPR round-trip),
// build the bf16 A fragment from fp32 x in registers, issue 16 WMMAs.
// 512 v_wmma_f32_16x16x32_bf16 per wave.
// ---------------------------------------------------------------------------
__global__ void __launch_bounds__(256) gemm_e_kernel(
    const float* __restrict__ x, const unsigned int* __restrict__ w1sw,
    const float* __restrict__ b1, const float* __restrict__ W2,
    const float* __restrict__ b2, float* __restrict__ e) {
    __shared__ uint4 smem[1024];                 // 16KB: one K-chunk of W1 frags

    const int tid  = threadIdx.x;
    const int lane = tid & 31;
    const int wave = tid >> 5;                   // 0..7
    const long R   = ((long)blockIdx.x * 8 + wave) * 16;  // row-tile base
    const int n    = lane & 15;                  // N index / A-row within tile
    const int hi   = (lane >> 4) & 1;            // which K half this lane holds

    v8f acc[16];
    const v8f vzero = {0.f,0.f,0.f,0.f,0.f,0.f,0.f,0.f};
#pragma unroll
    for (int i = 0; i < 16; ++i) acc[i] = vzero;

    const float* xbase = x + (R + n) * (long)D_DIM + hi * 8;

    // Per-thread 64B slice of the 16KB chunk: LDS dest + global source.
    // ISA: INST_OFFSET is added to BOTH the LDS and global addresses, so a
    // single (lds, global) pair + offset:0/16/32/48 covers the 4x b128 moves.
    const unsigned ldsdst = (unsigned)(uintptr_t)(&smem[tid * 4]);

    for (int kc = 0; kc < 32; ++kc) {
        __syncthreads();                         // previous chunk fully consumed
        {
            unsigned long long gaddr = (unsigned long long)(uintptr_t)
                ((const char*)w1sw + (size_t)kc * 16384 + (size_t)tid * 64);
            asm volatile(
                "global_load_async_to_lds_b128 %0, %1, off\n\t"
                "global_load_async_to_lds_b128 %0, %1, off offset:16\n\t"
                "global_load_async_to_lds_b128 %0, %1, off offset:32\n\t"
                "global_load_async_to_lds_b128 %0, %1, off offset:48\n\t"
                "s_wait_asynccnt 0x0"
                :: "v"(ldsdst), "v"(gaddr) : "memory");
        }
        __syncthreads();

        const float* xp = xbase + kc * 32;
        if (kc < 31) __builtin_prefetch(xp + 32, 0, 1);   // global_prefetch_b8

        // A fragment: 16 bf16 per lane; elems 0..7 = K kbase..+7,
        // elems 8..15 = K kbase+16..+23 (per ISA 16-bit A 16x32 layout).
        float4 c0 = *(const float4*)(xp + 0);
        float4 c1 = *(const float4*)(xp + 4);
        float4 c2 = *(const float4*)(xp + 16);
        float4 c3 = *(const float4*)(xp + 20);
        v16bf a;
        a[0]=(__bf16)c0.x;  a[1]=(__bf16)c0.y;  a[2]=(__bf16)c0.z;  a[3]=(__bf16)c0.w;
        a[4]=(__bf16)c1.x;  a[5]=(__bf16)c1.y;  a[6]=(__bf16)c1.z;  a[7]=(__bf16)c1.w;
        a[8]=(__bf16)c2.x;  a[9]=(__bf16)c2.y;  a[10]=(__bf16)c2.z; a[11]=(__bf16)c2.w;
        a[12]=(__bf16)c3.x; a[13]=(__bf16)c3.y; a[14]=(__bf16)c3.z; a[15]=(__bf16)c3.w;

#pragma unroll
        for (int nt = 0; nt < 16; ++nt) {
            union { uint4 q[2]; v16bf v; } u;
            u.q[0] = smem[nt * 64 + lane * 2];
            u.q[1] = smem[nt * 64 + lane * 2 + 1];
            acc[nt] = __builtin_amdgcn_wmma_f32_16x16x32_bf16(
                false, a, false, u.v, (short)0, acc[nt], false, false);
        }
    }

    // Epilogue: h = tanh(acc + b1); partial e per lane = sum_N h * W2[N].
    // C layout: lane<16 VGPR j -> (M=j, N=lane); lane>=16 -> (M=8+j, N=lane-16).
    const float b2v = b2[0];
    float ep[8];
#pragma unroll
    for (int j = 0; j < 8; ++j) ep[j] = 0.f;

#pragma unroll
    for (int nt = 0; nt < 16; ++nt) {
        float b1v = b1[nt * 16 + n];
        float w2v = W2[nt * 16 + n];
#pragma unroll
        for (int j = 0; j < 8; ++j)
            ep[j] += tanhf(acc[nt][j] + b1v) * w2v;
    }
    // Reduce over N: 16 lanes within each wave half (masks 1,2,4,8 stay in-half)
#pragma unroll
    for (int m = 1; m < 16; m <<= 1) {
#pragma unroll
        for (int j = 0; j < 8; ++j) ep[j] += __shfl_xor(ep[j], m, 32);
    }
    if (n == 0) {
        long rb = R + hi * 8;                    // lane0: M 0..7, lane16: M 8..15
#pragma unroll
        for (int j = 0; j < 8; ++j) e[rb + j] = ep[j] + b2v;
    }
}

// ---------------------------------------------------------------------------
// Kernel 3: exact order-statistic threshold per batch.
// threshold = sorted(e[b,:])[T/2]. Stable rank counting in LDS: exactly one
// element has rank == T/2, that element's value is the threshold.
// ---------------------------------------------------------------------------
__global__ void __launch_bounds__(256) thresh_kernel(
    const float* __restrict__ e, float* __restrict__ thr) {
    __shared__ float se[T_SEQ];                  // 16KB
    const int b = blockIdx.x;
    const float* eb = e + (long)b * T_SEQ;
    for (int i = threadIdx.x; i < T_SEQ; i += 256) se[i] = eb[i];
    __syncthreads();
    for (int i = threadIdx.x; i < T_SEQ; i += 256) {
        float v = se[i];
        int rank = 0;
        for (int j = 0; j < T_SEQ; ++j) {
            float w = se[j];
            rank += (w < v) || (w == v && j < i);
        }
        if (rank == T_SEQ / 2) thr[b] = v;
    }
}

// ---------------------------------------------------------------------------
// Kernel 4: masked softmax over T per batch -> beta.
// ---------------------------------------------------------------------------
__global__ void __launch_bounds__(256) softmax_kernel(
    const float* __restrict__ e, const float* __restrict__ thr,
    float* __restrict__ beta) {
    __shared__ float red[256];
    const int b = blockIdx.x;
    const float tv = thr[b];
    const float* eb = e + (long)b * T_SEQ;
    float* bb = beta + (long)b * T_SEQ;

    float mx = -3.4e38f;
    for (int i = threadIdx.x; i < T_SEQ; i += 256) {
        float v = eb[i];
        v += (v >= tv) ? NEG_BIG : 1.0f;
        mx = fmaxf(mx, v);
    }
    red[threadIdx.x] = mx;
    __syncthreads();
    for (int s = 128; s > 0; s >>= 1) {
        if (threadIdx.x < s) red[threadIdx.x] = fmaxf(red[threadIdx.x], red[threadIdx.x + s]);
        __syncthreads();
    }
    mx = red[0];
    __syncthreads();

    float sum = 0.f;
    for (int i = threadIdx.x; i < T_SEQ; i += 256) {
        float v = eb[i];
        v += (v >= tv) ? NEG_BIG : 1.0f;
        sum += __expf(v - mx);
    }
    red[threadIdx.x] = sum;
    __syncthreads();
    for (int s = 128; s > 0; s >>= 1) {
        if (threadIdx.x < s) red[threadIdx.x] += red[threadIdx.x + s];
        __syncthreads();
    }
    const float inv = 1.0f / red[0];

    for (int i = threadIdx.x; i < T_SEQ; i += 256) {
        float v = eb[i];
        v += (v >= tv) ? NEG_BIG : 1.0f;
        bb[i] = __expf(v - mx) * inv;
    }
}

// ---------------------------------------------------------------------------
// Kernel 5: out[b,d] = sum_t beta[b,t] * x[b,t,d]. Streams x (memory-bound),
// beta staged in LDS in 256-element chunks, float4 loads per thread.
// ---------------------------------------------------------------------------
__global__ void __launch_bounds__(256) wsum_kernel(
    const float* __restrict__ x, const float* __restrict__ beta,
    float* __restrict__ out) {
    __shared__ float sb[256];
    const int b  = blockIdx.x;
    const int d0 = threadIdx.x * 4;
    const float* xb = x + (long)b * T_SEQ * D_DIM;
    const float* bb = beta + (long)b * T_SEQ;

    float4 acc = make_float4(0.f, 0.f, 0.f, 0.f);
    for (int t0 = 0; t0 < T_SEQ; t0 += 256) {
        __syncthreads();
        sb[threadIdx.x] = bb[t0 + threadIdx.x];
        __syncthreads();
        for (int t = 0; t < 256; ++t) {
            float w = sb[t];
            const float4 xv = *(const float4*)(xb + (long)(t0 + t) * D_DIM + d0);
            acc.x += w * xv.x; acc.y += w * xv.y;
            acc.z += w * xv.z; acc.w += w * xv.w;
        }
    }
    *(float4*)(out + (long)b * D_DIM + d0) = acc;
}

// ---------------------------------------------------------------------------
extern "C" void kernel_launch(void* const* d_in, const int* in_sizes, int n_in,
                              void* d_out, int out_size, void* d_ws, size_t ws_size,
                              hipStream_t stream) {
    const float* x  = (const float*)d_in[0];   // (32, 4096, 1024)
    const float* W1 = (const float*)d_in[1];   // (1024, 256)
    const float* b1 = (const float*)d_in[2];   // (256,)
    const float* W2 = (const float*)d_in[3];   // (256, 1)
    const float* b2 = (const float*)d_in[4];   // (1,)
    float* out = (float*)d_out;                // (32, 1024, 1, 1)

    char* ws = (char*)d_ws;
    unsigned int* w1sw = (unsigned int*)ws;                    // 512 KB swizzled bf16 W1
    float* e    = (float*)(ws + (512 << 10));                  // 512 KB e[b,t]
    float* thr  = (float*)(ws + (1024 << 10));                 // 128 B thresholds
    float* beta = (float*)(ws + (1024 << 10) + 4096);          // 512 KB beta[b,t]

    conv_w1_kernel <<<64,   256, 0, stream>>>(W1, w1sw);
    gemm_e_kernel  <<<1024, 256, 0, stream>>>(x, w1sw, b1, W2, b2, e);
    thresh_kernel  <<<B_BATCH, 256, 0, stream>>>(e, thr);
    softmax_kernel <<<B_BATCH, 256, 0, stream>>>(e, thr, beta);
    wsum_kernel    <<<B_BATCH, 256, 0, stream>>>(x, beta, out);
}
